// ImprovedDeepGpcmCoralModel_27298812133854
// MI455X (gfx1250) — matrix operand order repaired and checked by reading, not compiled, not run on hardware
//
#include <hip/hip_runtime.h>
#include <hip/hip_bf16.h>

// ---------------------------------------------------------------------------
// ImprovedDeepGpcmCoralModel on MI455X (gfx1250, wave32, WMMA).
// Dims: B=128 S=1024 Q=5000 K=4 M=64 KD=128 VD=256 FD=128.
// v2: all WMMA fragments are 128-bit loadable.
//   - B matrices pre-swizzled once into fragment-major bf16 (global, L2-hot)
//   - A tiles staged into LDS in fragment order (aligned 16B chunks)
// ---------------------------------------------------------------------------

typedef __bf16 bf16;
typedef __attribute__((ext_vector_type(16))) __bf16 v16bf;
typedef __attribute__((ext_vector_type(8)))  __bf16 v8bf;
typedef __attribute__((ext_vector_type(8)))  float  v8f;

#define Bdim 128
#define Sdim 1024
#define Qn   5000
#define Mdim 64
#define KD   128
#define VD   256
#define FD   128
#define BSn  (Bdim * Sdim)   // 131072 rows

// Fragment layouts (ISA 7.12.2, wave32, 16-bit):
//  A 16x32: lane = m + 16*hi, hi=(Krel>>3)&1, e = (Krel&7)|((Krel>>1)&8)
//  B 32x16: lane = (n&15) + 16*hi, hi=(Krel>>3)&1, e = (Krel&7)|((Krel>>1)&8)
//  C  f32 : reg r -> M = r + (lane>=16 ? 8 : 0), N = lane&15

__device__ __forceinline__ v8f wmma_bf16(v16bf a, v16bf b, v8f c) {
  return __builtin_amdgcn_wmma_f32_16x16x32_bf16(false, a, false, b, (short)0, c, false, false);
}
__device__ __forceinline__ float sigf(float x) { return 1.f / (1.f + __expf(-x)); }

// ===========================================================================
// K0: pack a KdimxNdim f32 matrix (optionally stored transposed as NdimxKdim)
// into fragment-major bf16: dst[((ks*ntN+nt)*32+lane)*16+e].
// ===========================================================================
__global__ void k_pack_b(const float* __restrict__ src, bf16* __restrict__ dst,
                         int Kdim, int Ndim, int transposed) {
  const int total = Kdim * Ndim;
  const int ntN = Ndim >> 4;
  for (int p = blockIdx.x * blockDim.x + threadIdx.x; p < total;
       p += gridDim.x * blockDim.x) {
    const int e    = p & 15;
    const int lane = (p >> 4) & 31;
    const int tile = p >> 9;
    const int nt   = tile % ntN;
    const int ks   = tile / ntN;
    const int K    = ks * 32 + ((lane & 16) ? 8 : 0) + (e & 7) + ((e & 8) ? 16 : 0);
    const int n    = nt * 16 + (lane & 15);
    const float v  = transposed ? src[n * Kdim + K] : src[K * Ndim + n];
    dst[p] = (bf16)v;
  }
}

// ===========================================================================
// K1: correlation weight  w = softmax(qe @ key_memory^T)   (BS x 64), f32 out
// one wave per 16-row tile; A staged fragment-major, B from packed global.
// ===========================================================================
__global__ void k_corr_weight(const int* __restrict__ qd,
                              const float* __restrict__ table,
                              const bf16* __restrict__ kmp,   // packed 128x64
                              float* __restrict__ w_out) {
  __shared__ __align__(32) bf16 sA[4 * 512];   // 4 K-steps, fragment-major
  __shared__ float sL[16 * Mdim];
  const int lane = threadIdx.x;                // 32 threads
  const int row0 = blockIdx.x * 16;

  for (int ch = lane; ch < 256; ch += 32) {    // 16 rows x 16 chunks of 8 cols
    const int r = ch >> 4, g = ch & 15;
    const int c0 = g * 8;
    const int ks = c0 >> 5, rel = c0 & 31;
    const int hi = (rel >> 3) & 1, e0 = (rel & 7) | ((rel >> 1) & 8);
    const float* s = table + (size_t)qd[row0 + r] * KD + c0;
    v8bf v;
#pragma unroll
    for (int j = 0; j < 8; ++j) v[j] = (bf16)s[j];
    *(v8bf*)&sA[ks * 512 + (r + 16 * hi) * 16 + e0] = v;
  }
  __syncthreads();

  v8f acc[4];
#pragma unroll
  for (int nt = 0; nt < 4; ++nt)
#pragma unroll
    for (int j = 0; j < 8; ++j) acc[nt][j] = 0.f;

#pragma unroll
  for (int ks = 0; ks < 4; ++ks) {
    const v16bf a = *(const v16bf*)&sA[ks * 512 + lane * 16];
#pragma unroll
    for (int nt = 0; nt < 4; ++nt) {
      const v16bf b = *(const v16bf*)&kmp[((ks * 4 + nt) * 32 + lane) * 16];
      acc[nt] = wmma_bf16(a, b, acc[nt]);
    }
  }

#pragma unroll
  for (int nt = 0; nt < 4; ++nt)
#pragma unroll
    for (int r = 0; r < 8; ++r) {
      const int m = r + ((lane & 16) ? 8 : 0);
      sL[m * Mdim + nt * 16 + (lane & 15)] = acc[nt][r];
    }
  __syncthreads();

  if (lane < 16) {                              // row-wise softmax over 64
    float mx = -1e30f;
    for (int n = 0; n < Mdim; ++n) mx = fmaxf(mx, sL[lane * Mdim + n]);
    float s = 0.f;
    for (int n = 0; n < Mdim; ++n) { const float e = __expf(sL[lane * Mdim + n] - mx); sL[lane * Mdim + n] = e; s += e; }
    const float inv = 1.f / s;
    float* dst = w_out + (size_t)(row0 + lane) * Mdim;
    for (int n = 0; n < Mdim; ++n) dst[n] = sL[lane * Mdim + n] * inv;
  }
}

// ===========================================================================
// K2: ordered value embedding (one-hot GEMM collapsed to a 2-row gather).
// ===========================================================================
__global__ void k_value_embed(const int* __restrict__ qd, const int* __restrict__ rd,
                              const float* __restrict__ Wv, const float* __restrict__ bv,
                              bf16* __restrict__ ve) {
  const int flat = blockIdx.x;
  const int v = threadIdx.x;                   // 256
  const int q = qd[flat];
  const float scale = (float)rd[flat] * (1.f / 3.f);
  float x = bv[v];
  if (q > 0)
    x += Wv[(size_t)(q - 1) * VD + v] + scale * Wv[(size_t)(Qn + q - 1) * VD + v];
  ve[(size_t)flat * VD + v] = (bf16)x;
}

// ===========================================================================
// K3: erase = sigmoid(ve@We+be), add = tanh(ve@Wa+ba). blockIdx.y selects.
// 256 threads = 8 waves: 4 row-tiles x 2 col-halves; 64x256 per CTA.
// ===========================================================================
__global__ void k_erase_add(const bf16* __restrict__ ve,
                            const bf16* __restrict__ wep, const float* __restrict__ be,
                            const bf16* __restrict__ wap, const float* __restrict__ ba,
                            bf16* __restrict__ eo, bf16* __restrict__ ao) {
  __shared__ __align__(32) bf16 sA[4 * 512];   // 4 row-tiles, fragment-major
  const bool isAdd = (blockIdx.y != 0);
  const bf16* Wp    = isAdd ? wap : wep;
  const float* bias = isAdd ? ba : be;
  bf16* out         = isAdd ? ao : eo;
  const int row0 = blockIdx.x * 64;
  const int tid = threadIdx.x, lane = tid & 31, wave = tid >> 5;
  const int wrt = wave & 3;       // row tile
  const int wch = wave >> 2;      // col half (128 cols)

  v8f acc[8];
#pragma unroll
  for (int nt = 0; nt < 8; ++nt)
#pragma unroll
    for (int j = 0; j < 8; ++j) acc[nt][j] = 0.f;

  for (int ks = 0; ks < 8; ++ks) {
    const int k0 = ks * 32;
    {                                           // stage A: one 16B chunk/thread
      const int r = tid >> 2, c0 = (tid & 3) * 8;
      const int hi = (c0 >> 3) & 1, e0 = (c0 & 7) | ((c0 >> 1) & 8);
      const int rt = r >> 4, m = r & 15;
      const v8bf v = *(const v8bf*)&ve[(size_t)(row0 + r) * VD + k0 + c0];
      *(v8bf*)&sA[rt * 512 + (m + 16 * hi) * 16 + e0] = v;
    }
    __syncthreads();
    const v16bf a = *(const v16bf*)&sA[wrt * 512 + lane * 16];
#pragma unroll
    for (int nt = 0; nt < 8; ++nt) {
      const v16bf b = *(const v16bf*)&Wp[((size_t)(ks * 16 + wch * 8 + nt) * 32 + lane) * 16];
      acc[nt] = wmma_bf16(a, b, acc[nt]);
    }
    __syncthreads();
  }

#pragma unroll
  for (int nt = 0; nt < 8; ++nt)
#pragma unroll
    for (int r = 0; r < 8; ++r) {
      const int m = r + ((lane & 16) ? 8 : 0);
      const int n = wch * 128 + nt * 16 + (lane & 15);
      float x = acc[nt][r] + bias[n];
      x = isAdd ? tanhf(x) : sigf(x);
      out[(size_t)(row0 + wrt * 16 + m) * VD + n] = (bf16)x;
    }
}

// ===========================================================================
// K4: recurrent scan. One CTA per batch row b; thread v owns Mv[:,v] (64 f32
// in VGPRs). Per step: broadcast w_t via LDS, 3 FMAs/slot, emit read_t.
// ===========================================================================
__global__ void k_scan(const float* __restrict__ w_ws,
                       const bf16* __restrict__ e_ws, const bf16* __restrict__ a_ws,
                       const float* __restrict__ Mv0, bf16* __restrict__ read_out) {
  __shared__ float sw[Mdim];
  const int b = blockIdx.x;
  const int v = threadIdx.x;                   // 256
  float Mv[Mdim];
#pragma unroll
  for (int m = 0; m < Mdim; ++m) Mv[m] = Mv0[m * VD + v];

  const float* wrow = w_ws + (size_t)b * Sdim * Mdim;
  const bf16*  erow = e_ws + (size_t)b * Sdim * VD;
  const bf16*  arow = a_ws + (size_t)b * Sdim * VD;
  bf16*        rrow = read_out + (size_t)b * Sdim * VD;

  for (int t = 0; t < Sdim; ++t) {
    if (v < Mdim) sw[v] = wrow[t * Mdim + v];
    __syncthreads();
    const float e = (float)erow[t * VD + v];
    const float a = (float)arow[t * VD + v];
    if (t + 1 < Sdim) {                        // stream next step into cache
      __builtin_prefetch(&erow[(t + 1) * VD + v], 0, 0);
      __builtin_prefetch(&arow[(t + 1) * VD + v], 0, 0);
    }
    float read = 0.f;
#pragma unroll
    for (int m = 0; m < Mdim; ++m) {
      const float wm = sw[m];
      read += wm * Mv[m];
      Mv[m] = Mv[m] * (1.f - wm * e) + wm * a;
    }
    rrow[t * VD + v] = (bf16)read;
    __syncthreads();
  }
}

// ===========================================================================
// K5: summary GEMM (16x384x128) + IRT/CORAL head + output scatter.
// 128 threads (4 waves) per 16-row tile; each wave owns 2 N-tiles.
// ===========================================================================
__global__ void k_head(const int* __restrict__ qd, const float* __restrict__ table,
                       const bf16* __restrict__ read_ws,
                       const bf16* __restrict__ wsp, const float* __restrict__ bs,
                       const float* __restrict__ Wth, const float* __restrict__ bth,
                       const float* __restrict__ Wbe, const float* __restrict__ bbe,
                       const float* __restrict__ Wd,  const float* __restrict__ bd,
                       const float* __restrict__ Wc1, const float* __restrict__ bc1,
                       const float* __restrict__ Wc2, const float* __restrict__ bc2,
                       const float* __restrict__ cw,  const float* __restrict__ cb,
                       float* __restrict__ out) {
  __shared__ __align__(32) bf16 sA[12 * 512];  // X=[read|qe] fragment-major
  __shared__ float sS[16 * FD];                // summary
  __shared__ float sQ[16 * KD];                // qe f32 (for betas/disc)
  const int row0 = blockIdx.x * 16;
  const int tid = threadIdx.x, lane = tid & 31, wave = tid >> 5;

  for (int ch = tid; ch < 768; ch += 128) {    // 16 rows x 48 chunks of 8
    const int r = ch / 48, g = ch % 48;
    const int c0 = g * 8;
    const int ks = c0 >> 5, rel = c0 & 31;
    const int hi = (rel >> 3) & 1, e0 = (rel & 7) | ((rel >> 1) & 8);
    v8bf v;
    if (c0 < VD) {
      v = *(const v8bf*)&read_ws[(size_t)(row0 + r) * VD + c0];
    } else {
      const float* s = table + (size_t)qd[row0 + r] * KD + (c0 - VD);
#pragma unroll
      for (int j = 0; j < 8; ++j) { const float f = s[j]; v[j] = (bf16)f; sQ[r * KD + (c0 - VD) + j] = f; }
    }
    *(v8bf*)&sA[ks * 512 + (r + 16 * hi) * 16 + e0] = v;
  }
  __syncthreads();

  v8f acc[2];
#pragma unroll
  for (int nt = 0; nt < 2; ++nt)
#pragma unroll
    for (int j = 0; j < 8; ++j) acc[nt][j] = 0.f;

  for (int ks = 0; ks < 12; ++ks) {
    const v16bf a = *(const v16bf*)&sA[ks * 512 + lane * 16];
#pragma unroll
    for (int nt = 0; nt < 2; ++nt) {
      const v16bf b = *(const v16bf*)&wsp[((ks * 8 + wave * 2 + nt) * 32 + lane) * 16];
      acc[nt] = wmma_bf16(a, b, acc[nt]);
    }
  }
#pragma unroll
  for (int nt = 0; nt < 2; ++nt)
#pragma unroll
    for (int r = 0; r < 8; ++r) {
      const int m = r + ((lane & 16) ? 8 : 0);
      const int n = wave * 32 + nt * 16 + (lane & 15);
      sS[m * FD + n] = tanhf(acc[nt][r] + bs[n]);
    }
  __syncthreads();

  if (tid < 16) {                              // per-row scalar head
    const int m = tid;
    const size_t idx = (size_t)row0 + m;       // flat b*S+t (output-major)
    float th = 0.f, dd = 0.f;
    for (int j = 0; j < FD; ++j) { const float s = sS[m * FD + j]; th += s * Wth[j]; dd += s * Wd[j]; }
    th = (th + bth[0]) * 3.0f;                 // ABILITY_SCALE
    float bb[3] = {bbe[0], bbe[1], bbe[2]};
    for (int j = 0; j < KD; ++j) {
      const float qe = sQ[m * KD + j];
      dd += qe * Wd[FD + j];
      bb[0] += qe * Wbe[j * 3 + 0];
      bb[1] += qe * Wbe[j * 3 + 1];
      bb[2] += qe * Wbe[j * 3 + 2];
    }
    dd += bd[0];
    const float alpha = log1pf(__expf(dd));    // softplus
    const float feat[5] = {th, alpha, bb[0], bb[1], bb[2]};
    float h1[64];
    for (int j = 0; j < 64; ++j) {
      float x = bc1[j];
#pragma unroll
      for (int i2 = 0; i2 < 5; ++i2) x += feat[i2] * Wc1[i2 * 64 + j];
      h1[j] = fmaxf(x, 0.f);
    }
    float sdot = 0.f;
    for (int j = 0; j < 32; ++j) {
      float x = bc2[j];
      for (int i2 = 0; i2 < 64; ++i2) x += h1[i2] * Wc2[i2 * 32 + j];
      sdot += fmaxf(x, 0.f) * cw[j];
    }
    float lg[3], cum[3], c = 1.f;
#pragma unroll
    for (int k = 0; k < 3; ++k) { lg[k] = sdot + cb[k]; c *= sigf(lg[k]); cum[k] = c; }
    const float probs[4] = {1.f - cum[0], cum[0] - cum[1], cum[1] - cum[2], cum[2]};
    const size_t BS = BSn;
    out[idx] = th;                                               // thetas
    for (int k = 0; k < 3; ++k) out[BS + idx * 3 + k] = bb[k];   // betas
    out[4 * BS + idx] = alpha;                                   // alphas
    for (int k = 0; k < 4; ++k) out[5 * BS + idx * 4 + k] = probs[k];
    for (int k = 0; k < 3; ++k) out[9 * BS + idx * 3 + k] = lg[k];
  }
}

// ===========================================================================
extern "C" void kernel_launch(void* const* d_in, const int* in_sizes, int n_in,
                              void* d_out, int out_size, void* d_ws, size_t ws_size,
                              hipStream_t stream) {
  const int*   qd    = (const int*)d_in[0];
  const int*   rd    = (const int*)d_in[1];
  const float* table = (const float*)d_in[2];
  const float* km    = (const float*)d_in[3];
  const float* Mv0   = (const float*)d_in[4];
  const float* Wv    = (const float*)d_in[5];
  const float* bv    = (const float*)d_in[6];
  const float* We    = (const float*)d_in[7];
  const float* be    = (const float*)d_in[8];
  const float* Wa    = (const float*)d_in[9];
  const float* ba    = (const float*)d_in[10];
  const float* Ws    = (const float*)d_in[11];
  const float* bs    = (const float*)d_in[12];
  const float* Wth   = (const float*)d_in[13];
  const float* bth   = (const float*)d_in[14];
  const float* Wbe   = (const float*)d_in[15];
  const float* bbe   = (const float*)d_in[16];
  const float* Wd    = (const float*)d_in[17];
  const float* bd    = (const float*)d_in[18];
  const float* Wc1   = (const float*)d_in[19];
  const float* bc1   = (const float*)d_in[20];
  const float* Wc2   = (const float*)d_in[21];
  const float* bc2   = (const float*)d_in[22];
  const float* cw    = (const float*)d_in[23];
  const float* cb    = (const float*)d_in[24];
  float* out = (float*)d_out;

  // Workspace carve-up (~289 MB; every buffer fully written before read).
  char* p = (char*)d_ws;
  float* w_ws   = (float*)p; p += (size_t)BSn * Mdim * sizeof(float);  // 32 MB
  bf16* ve_ws   = (bf16*)p;  p += (size_t)BSn * VD * sizeof(bf16);     // 64 MB
  bf16* e_ws    = (bf16*)p;  p += (size_t)BSn * VD * sizeof(bf16);     // 64 MB
  bf16* a_ws    = (bf16*)p;  p += (size_t)BSn * VD * sizeof(bf16);     // 64 MB
  bf16* read_ws = (bf16*)p;  p += (size_t)BSn * VD * sizeof(bf16);     // 64 MB
  bf16* kmp     = (bf16*)p;  p += (size_t)KD * Mdim * sizeof(bf16);    // packed key^T
  bf16* wep     = (bf16*)p;  p += (size_t)VD * VD * sizeof(bf16);      // packed W_erase
  bf16* wap     = (bf16*)p;  p += (size_t)VD * VD * sizeof(bf16);      // packed W_add
  bf16* wsp     = (bf16*)p;  p += (size_t)(VD + KD) * FD * sizeof(bf16); // packed W_summary

  // Pre-swizzle B matrices into WMMA fragment-major layout (one-time, tiny).
  k_pack_b<<<64, 256, 0, stream>>>(km, kmp, KD, Mdim, 1);     // key_memory^T
  k_pack_b<<<256, 256, 0, stream>>>(We, wep, VD, VD, 0);
  k_pack_b<<<256, 256, 0, stream>>>(Wa, wap, VD, VD, 0);
  k_pack_b<<<256, 256, 0, stream>>>(Ws, wsp, VD + KD, FD, 0);

  k_corr_weight<<<BSn / 16, 32, 0, stream>>>(qd, table, kmp, w_ws);
  k_value_embed<<<BSn, VD, 0, stream>>>(qd, rd, Wv, bv, ve_ws);
  dim3 g3(BSn / 64, 2);
  k_erase_add<<<g3, 256, 0, stream>>>(ve_ws, wep, be, wap, ba, e_ws, a_ws);
  k_scan<<<Bdim, VD, 0, stream>>>(w_ws, e_ws, a_ws, Mv0, read_ws);
  k_head<<<BSn / 16, 128, 0, stream>>>(qd, table, read_ws, wsp, bs, Wth, bth,
                                       Wbe, bbe, Wd, bd, Wc1, bc1, Wc2, bc2,
                                       cw, cb, out);
}
